// PairUpdateWithAxialAttentionLayer_19344532701522
// MI455X (gfx1250) — matrix operand order repaired and checked by reading, not compiled, not run on hardware
//
#include <hip/hip_runtime.h>
#include <hip/hip_bf16.h>

// ---------------------------------------------------------------------------
// PairUpdateWithAxialAttentionLayer for MI455X (gfx1250, wave32, WMMA).
// All GEMMs use v_wmma_f32_16x16x32_f16 (f16 A/B, f32 accum).
// Weights are pre-transposed into f16 workspace so B-fragments are contiguous
// global_load_b128 pairs. Template-specialized stages keep kernarg pointer
// provenance so loads lower to global_* (not flat_*).
// ---------------------------------------------------------------------------

#define DEV static __device__ __forceinline__

typedef _Float16 h16;
typedef __attribute__((ext_vector_type(4)))  _Float16 v4h;
typedef __attribute__((ext_vector_type(8)))  _Float16 v8h;
typedef __attribute__((ext_vector_type(16))) _Float16 v16h;
typedef __attribute__((ext_vector_type(8)))  float    v8f;

constexpr int L    = 384;
constexpr int D    = 128;
constexpr int DFF  = 512;
constexpr int NTOK = L * L;                 // 147456 tokens
constexpr float NRM  = 0.4204482076268573f; // 32^(-0.25)
constexpr float EPSK = 1e-3f;

DEV v8f vzero8() {
  v8f z;
#pragma unroll
  for (int i = 0; i < 8; ++i) z[i] = 0.f;
  return z;
}

// A-fragment (16x32 f16, M=lane&15): 16 contiguous halves {kbase..+15}.
DEV v16h ld_a(const h16* p) {
  return *(const v16h*)p;
}
// B-fragment from a transposed (BT = [N][K]) matrix: 16 contiguous halves.
DEV v16h ld_b(const h16* p) { return *(const v16h*)p; }

DEV v8f wmma16(v16h a, v16h b, v8f c) {
  return __builtin_amdgcn_wmma_f32_16x16x32_f16(false, a, false, b,
                                                (short)0, c, false, false);
}

// ---------------------------------------------------------------------------
// Weight conversion: f32 [R,C] -> f16 transposed [C][R]
__global__ void k_cvtT(const float* __restrict__ src, h16* __restrict__ dst,
                       int R, int C) {
  int i = blockIdx.x * blockDim.x + threadIdx.x;
  if (i >= R * C) return;
  int r = i / C, c = i % C;
  dst[c * R + r] = (h16)src[i];
}
// Plain f32 -> f16 (for proj: [F][DH] is already the BT layout we need)
__global__ void k_cvt(const float* __restrict__ src, h16* __restrict__ dst, int n) {
  int i = blockIdx.x * blockDim.x + threadIdx.x;
  if (i < n) dst[i] = (h16)src[i];
}

// ---------------------------------------------------------------------------
// LayerNorm: one wave per token (D=128 -> 4 floats/lane), writes f16.
// Output is sequence-major: out[(s*384+t)*128 + d]; input addressed via strides
// so the row/col transposes are free.
__global__ void __launch_bounds__(256)
k_ln(const float* __restrict__ x, const float* __restrict__ g,
     const float* __restrict__ bta, h16* __restrict__ out,
     long strideT, long strideS) {
  int wid = threadIdx.x >> 5, lane = threadIdx.x & 31;
  int tk = blockIdx.x * 8 + wid;
  int s = tk / 384, t = tk % 384;
  long base = (long)t * strideT + (long)s * strideS + lane * 4;
  float4 xv = *(const float4*)(x + base);
  float sum = xv.x + xv.y + xv.z + xv.w;
  for (int o = 16; o > 0; o >>= 1) sum += __shfl_xor(sum, o, 32);
  float mu = sum * (1.f / 128.f);
  float d0 = xv.x - mu, d1 = xv.y - mu, d2 = xv.z - mu, d3 = xv.w - mu;
  float vs = d0 * d0 + d1 * d1 + d2 * d2 + d3 * d3;
  for (int o = 16; o > 0; o >>= 1) vs += __shfl_xor(vs, o, 32);
  float rs = rsqrtf(vs * (1.f / 128.f) + 1e-5f);
  int c = lane * 4;
  v4h o4;
  o4[0] = (h16)(d0 * rs * g[c + 0] + bta[c + 0]);
  o4[1] = (h16)(d1 * rs * g[c + 1] + bta[c + 1]);
  o4[2] = (h16)(d2 * rs * g[c + 2] + bta[c + 2]);
  o4[3] = (h16)(d3 * rs * g[c + 3] + bta[c + 3]);
  *(v4h*)(out + (long)tk * 128 + c) = o4;
}

// ---------------------------------------------------------------------------
// One stage (q / k / v) of the fused QKV + performer feature map.
// MODE: 0 = q (scale, project, relu+eps, row-major qp)
//       1 = k (scale, project, relu+eps, transposed kp)
//       2 = v (bias only, transposed store)
// Weight/bias pointers arrive directly from kernel arguments so address-space
// inference keeps every load on the global path.
template <int MODE>
DEV void qkv_stage(const h16* __restrict__ W, const float* __restrict__ bb,
                   const h16* __restrict__ proj16,
                   const v16h (&ah)[4], const v16h (&pb)[4],
                   h16* sQw, int s, int t0, int m, int hi,
                   h16* __restrict__ qp, h16* __restrict__ kp,
                   h16* __restrict__ vt) {
  for (int nt = 0; nt < 8; ++nt) {
    v8f acc = vzero8();
#pragma unroll
    for (int kk = 0; kk < 4; ++kk)
      acc = wmma16(ah[kk], ld_b(W + (nt * 16 + m) * 128 + kk * 32 + hi * 16), acc);
    int col = nt * 16 + m;
    float bias = bb[col];
    if (MODE == 2) {                  // v: write transposed [dh][n], contiguous n
      v8h pk;
#pragma unroll
      for (int r = 0; r < 8; ++r) pk[r] = (h16)(acc[r] + bias);
      int head = col >> 5, dh = col & 31;
      *(v8h*)(vt + (((long)(s * 4 + head) * 32 + dh) * 384 + t0 + hi * 8)) = pk;
    } else {                          // q/k: stage to LDS in row-major f16
#pragma unroll
      for (int r = 0; r < 8; ++r)
        sQw[(r + 8 * hi) * 128 + col] = (h16)((acc[r] + bias) * NRM);
    }
  }
  if (MODE < 2) {                     // performer feature map per head
    for (int head = 0; head < 4; ++head) {
      v16h aq = ld_a(&sQw[m * 128 + head * 32 + hi * 16]);
#pragma unroll
      for (int ft = 0; ft < 4; ++ft) {
        v8f acc = vzero8();
        acc = wmma16(aq, pb[ft], acc);
        int f = ft * 16 + m;
        if (MODE == 0) {              // qp row-major [n][64]
#pragma unroll
          for (int r = 0; r < 8; ++r) {
            int n = t0 + r + 8 * hi;
            qp[((long)(s * 4 + head) * 384 + n) * 64 + f] =
                (h16)(fmaxf(acc[r], 0.f) + EPSK);
          }
        } else {                      // kp transposed [f][n], contiguous n
          v8h pk;
#pragma unroll
          for (int r = 0; r < 8; ++r) pk[r] = (h16)(fmaxf(acc[r], 0.f) + EPSK);
          *(v8h*)(kp + (((long)(s * 4 + head) * 64 + f) * 384 + t0 + hi * 8)) = pk;
        }
      }
    }
  }
  (void)proj16;
}

// grid (seq=384, 6); block 128 (4 waves); wave handles a 16-token tile.
__global__ void __launch_bounds__(128)
k_qkvp(const h16* __restrict__ hseq,
       const h16* __restrict__ WqT, const h16* __restrict__ WkT,
       const h16* __restrict__ WvT,
       const float* __restrict__ bq, const float* __restrict__ bk,
       const float* __restrict__ bv,
       const h16* __restrict__ proj16,
       h16* __restrict__ qp, h16* __restrict__ kp, h16* __restrict__ vt) {
  __shared__ alignas(32) h16 sQ[4][16 * 128];
  int s = blockIdx.x;
  int w = threadIdx.x >> 5, lane = threadIdx.x & 31;
  int m = lane & 15, hi = lane >> 4;
  int t0 = blockIdx.y * 64 + w * 16;
  const h16* hrow = hseq + ((long)s * 384 + t0) * 128;

  v16h ah[4];
#pragma unroll
  for (int kk = 0; kk < 4; ++kk)
    ah[kk] = ld_a(hrow + m * 128 + kk * 32 + hi * 16);

  // proj B-fragments are invariant across heads and q/k passes: hoist.
  v16h pb[4];
#pragma unroll
  for (int ft = 0; ft < 4; ++ft)
    pb[ft] = ld_b(proj16 + (ft * 16 + m) * 32 + hi * 16);

  qkv_stage<0>(WqT, bq, proj16, ah, pb, sQ[w], s, t0, m, hi, qp, kp, vt);
  qkv_stage<1>(WkT, bk, proj16, ah, pb, sQ[w], s, t0, m, hi, qp, kp, vt);
  qkv_stage<2>(WvT, bv, proj16, ah, pb, sQ[w], s, t0, m, hi, qp, kp, vt);
}

// ---------------------------------------------------------------------------
// Performer core per (seq, head): ksum, dinv, ctx = kp^T @ v, out = qp@ctx*dinv
// grid (384, 4); block 128 (4 waves).
__global__ void __launch_bounds__(128)
k_perf(const h16* __restrict__ qp_all, const h16* __restrict__ kp_all,
       const h16* __restrict__ vt_all, h16* __restrict__ attn,
       long sT, long sS) {
  __shared__ float sKsum[64];
  __shared__ float sDinv[384];
  __shared__ alignas(32) h16 sCtx[32 * 64];   // ctx^T : [dh][f]
  int s = blockIdx.x, hd = blockIdx.y;
  long off = (long)(s * 4 + hd);
  const h16* qp  = qp_all + off * 384 * 64;
  const h16* kpT = kp_all + off * 64 * 384;
  const h16* vT  = vt_all + off * 32 * 384;
  int tid = threadIdx.x;
  int w = tid >> 5, lane = tid & 31, m = lane & 15, hi = lane >> 4;

  if (tid < 64) {                       // ksum[f] = sum_n kp[f][n]
    const v8h* row = (const v8h*)(kpT + tid * 384);
    float a = 0.f;
    for (int i = 0; i < 48; ++i) {
      v8h c = row[i];
#pragma unroll
      for (int j = 0; j < 8; ++j) a += (float)c[j];
    }
    sKsum[tid] = a;
  }
  __syncthreads();

  for (int jj = 0; jj < 3; ++jj) {      // dinv[n] = 1 / (qp[n,:] . ksum)
    int n = tid + jj * 128;
    const v8h* row = (const v8h*)(qp + (long)n * 64);
    float a = 0.f;
#pragma unroll
    for (int i = 0; i < 8; ++i) {
      v8h c = row[i];
#pragma unroll
      for (int j = 0; j < 8; ++j) a += (float)c[j] * sKsum[i * 8 + j];
    }
    sDinv[n] = 1.f / a;
  }

  // ctx = kp^T (64x384) @ v (384x32): wave w owns f-tile w; K = 12 x 32
#pragma unroll
  for (int nt = 0; nt < 2; ++nt) {
    v8f acc = vzero8();
    for (int kk = 0; kk < 12; ++kk)
      acc = wmma16(ld_a(kpT + (w * 16 + m) * 384 + kk * 32 + hi * 16),
                   ld_b(vT + (nt * 16 + m) * 384 + kk * 32 + hi * 16), acc);
#pragma unroll
    for (int r = 0; r < 8; ++r)
      sCtx[(nt * 16 + m) * 64 + (w * 16 + r + 8 * hi)] = (h16)acc[r];
  }
  __syncthreads();

  // out = qp (384x64) @ ctx (64x32), scaled by dinv; scatter token-major f16
  for (int i = 0; i < 6; ++i) {
    int mt = i * 4 + w;
#pragma unroll
    for (int nt = 0; nt < 2; ++nt) {
      v8f acc = vzero8();
#pragma unroll
      for (int kk = 0; kk < 2; ++kk)
        acc = wmma16(ld_a(qp + (mt * 16 + m) * 64 + kk * 32 + hi * 16),
                     ld_b(&sCtx[(nt * 16 + m) * 64 + kk * 32 + hi * 16]), acc);
#pragma unroll
      for (int r = 0; r < 8; ++r) {
        int n = mt * 16 + r + 8 * hi;
        long tok = (long)n * sT + (long)s * sS;
        attn[tok * 128 + hd * 32 + nt * 16 + m] = (h16)(acc[r] * sDinv[n]);
      }
    }
  }
}

// ---------------------------------------------------------------------------
// Output projection + bias + residual -> f32.  grid NTOK/64; block 128.
__global__ void __launch_bounds__(128)
k_wo(const h16* __restrict__ attn, const h16* __restrict__ WoT,
     const float* __restrict__ bo, const float* __restrict__ resid,
     float* __restrict__ out) {
  int w = threadIdx.x >> 5, lane = threadIdx.x & 31, m = lane & 15, hi = lane >> 4;
  long t0 = (long)blockIdx.x * 64 + w * 16;
  const h16* arow = attn + t0 * 128;
  v16h ah[4];
#pragma unroll
  for (int kk = 0; kk < 4; ++kk)
    ah[kk] = ld_a(arow + m * 128 + kk * 32 + hi * 16);
  for (int nt = 0; nt < 8; ++nt) {
    v8f acc = vzero8();
#pragma unroll
    for (int kk = 0; kk < 4; ++kk)
      acc = wmma16(ah[kk], ld_b(WoT + (nt * 16 + m) * 128 + kk * 32 + hi * 16), acc);
    int col = nt * 16 + m;
    float bias = bo[col];
#pragma unroll
    for (int r = 0; r < 8; ++r) {
      long tok = t0 + r + 8 * hi;
      out[tok * 128 + col] = resid[tok * 128 + col] + acc[r] + bias;
    }
  }
}

// ---------------------------------------------------------------------------
// Fused FFN: relu(h@W1+b1)@W2+b2 + residual. [16,512] intermediate in LDS.
// grid NTOK/32; block 64 (2 waves).
__global__ void __launch_bounds__(64)
k_ffn(const h16* __restrict__ hh, const h16* __restrict__ W1T,
      const float* __restrict__ b1, const h16* __restrict__ W2T,
      const float* __restrict__ b2, const float* __restrict__ resid,
      float* __restrict__ out) {
  __shared__ alignas(32) h16 sU[2][16 * 512];
  int w = threadIdx.x >> 5, lane = threadIdx.x & 31, m = lane & 15, hi = lane >> 4;
  long t0 = (long)blockIdx.x * 32 + w * 16;
  const h16* hrow = hh + t0 * 128;
  v16h ah[4];
#pragma unroll
  for (int kk = 0; kk < 4; ++kk)
    ah[kk] = ld_a(hrow + m * 128 + kk * 32 + hi * 16);

  for (int nt = 0; nt < 32; ++nt) {            // u = relu(h @ W1 + b1)
    v8f acc = vzero8();
#pragma unroll
    for (int kk = 0; kk < 4; ++kk)
      acc = wmma16(ah[kk], ld_b(W1T + (nt * 16 + m) * 128 + kk * 32 + hi * 16), acc);
    int col = nt * 16 + m;
    float bias = b1[col];
#pragma unroll
    for (int r = 0; r < 8; ++r)
      sU[w][(r + 8 * hi) * 512 + col] = (h16)fmaxf(acc[r] + bias, 0.f);
  }
  // LDS region is wave-private; per-wave DS ordering suffices.
  for (int nt = 0; nt < 8; ++nt) {             // out = x2 + u @ W2 + b2
    v8f acc = vzero8();
    for (int kk = 0; kk < 16; ++kk)
      acc = wmma16(ld_a(&sU[w][m * 512 + kk * 32 + hi * 16]),
                   ld_b(W2T + (nt * 16 + m) * 512 + kk * 32 + hi * 16), acc);
    int col = nt * 16 + m;
    float bias = b2[col];
#pragma unroll
    for (int r = 0; r < 8; ++r) {
      long tok = t0 + r + 8 * hi;
      out[tok * 128 + col] = resid[tok * 128 + col] + acc[r] + bias;
    }
  }
}

// ---------------------------------------------------------------------------
extern "C" void kernel_launch(void* const* d_in, const int* in_sizes, int n_in,
                              void* d_out, int out_size, void* d_ws, size_t ws_size,
                              hipStream_t stream) {
  (void)in_sizes; (void)n_in; (void)out_size; (void)ws_size;
  const float* x   = (const float*)d_in[0];
  const float* rWq = (const float*)d_in[1];  const float* rbq = (const float*)d_in[2];
  const float* rWk = (const float*)d_in[3];  const float* rbk = (const float*)d_in[4];
  const float* rWv = (const float*)d_in[5];  const float* rbv = (const float*)d_in[6];
  const float* rWo = (const float*)d_in[7];  const float* rbo = (const float*)d_in[8];
  const float* rPj = (const float*)d_in[9];
  const float* cWq = (const float*)d_in[10]; const float* cbq = (const float*)d_in[11];
  const float* cWk = (const float*)d_in[12]; const float* cbk = (const float*)d_in[13];
  const float* cWv = (const float*)d_in[14]; const float* cbv = (const float*)d_in[15];
  const float* cWo = (const float*)d_in[16]; const float* cbo = (const float*)d_in[17];
  const float* cPj = (const float*)d_in[18];
  const float* ln1g = (const float*)d_in[19]; const float* ln1b = (const float*)d_in[20];
  const float* ln2g = (const float*)d_in[21]; const float* ln2b = (const float*)d_in[22];
  const float* ln3g = (const float*)d_in[23]; const float* ln3b = (const float*)d_in[24];
  const float* W1 = (const float*)d_in[25]; const float* fb1 = (const float*)d_in[26];
  const float* W2 = (const float*)d_in[27]; const float* fb2 = (const float*)d_in[28];
  float* outp = (float*)d_out;

  char* p = (char*)d_ws;
  auto bump = [&](size_t bytes) -> void* {
    void* q = (void*)p;
    p += (bytes + 255) & ~(size_t)255;
    return q;
  };
  h16* rWqT = (h16*)bump(128 * 128 * 2);
  h16* rWkT = (h16*)bump(128 * 128 * 2);
  h16* rWvT = (h16*)bump(128 * 128 * 2);
  h16* rWoT = (h16*)bump(128 * 128 * 2);
  h16* rPj16 = (h16*)bump(64 * 32 * 2);
  h16* cWqT = (h16*)bump(128 * 128 * 2);
  h16* cWkT = (h16*)bump(128 * 128 * 2);
  h16* cWvT = (h16*)bump(128 * 128 * 2);
  h16* cWoT = (h16*)bump(128 * 128 * 2);
  h16* cPj16 = (h16*)bump(64 * 32 * 2);
  h16* W1T = (h16*)bump((size_t)DFF * 128 * 2);  // [512][128]
  h16* W2T = (h16*)bump((size_t)DFF * 128 * 2);  // [128][512]
  h16* bufH  = (h16*)bump((size_t)NTOK * 128 * 2);    // LN out / attn (aliased)
  h16* bufV  = (h16*)bump((size_t)NTOK * 128 * 2);    // v transposed
  h16* bufQp = (h16*)bump((size_t)NTOK * 256 * 2);    // qp [s][h][n][64]
  h16* bufKp = (h16*)bump((size_t)NTOK * 256 * 2);    // kp [s][h][64][n]
  float* bufX1 = (float*)bump((size_t)NTOK * 128 * 4);
  h16* bufAttn = bufH;  // alias: LN output fully consumed before attn writes

  auto cvtT = [&](const float* s, h16* dd, int R, int C) {
    int n = R * C;
    k_cvtT<<<(n + 255) / 256, 256, 0, stream>>>(s, dd, R, C);
  };
  cvtT(rWq, rWqT, 128, 128); cvtT(rWk, rWkT, 128, 128);
  cvtT(rWv, rWvT, 128, 128); cvtT(rWo, rWoT, 128, 128);
  cvtT(cWq, cWqT, 128, 128); cvtT(cWk, cWkT, 128, 128);
  cvtT(cWv, cWvT, 128, 128); cvtT(cWo, cWoT, 128, 128);
  cvtT(W1, W1T, 128, 512);   cvtT(W2, W2T, 512, 128);
  k_cvt<<<(64 * 32 + 255) / 256, 256, 0, stream>>>(rPj, rPj16, 64 * 32);
  k_cvt<<<(64 * 32 + 255) / 256, 256, 0, stream>>>(cPj, cPj16, 64 * 32);

  const long SD = (long)L * D;  // 49152

  // ---- Row-wise attention (sequences along n; seq index = l) ----
  k_ln<<<NTOK / 8, 256, 0, stream>>>(x, ln1g, ln1b, bufH, SD, 128);
  k_qkvp<<<dim3(L, 6), 128, 0, stream>>>(bufH, rWqT, rWkT, rWvT, rbq, rbk, rbv,
                                         rPj16, bufQp, bufKp, bufV);
  k_perf<<<dim3(L, 4), 128, 0, stream>>>(bufQp, bufKp, bufV, bufAttn, 384, 1);
  k_wo<<<NTOK / 64, 128, 0, stream>>>(bufAttn, rWoT, rbo, x, bufX1);

  // ---- Col-wise attention (sequences along l; seq index = n) ----
  k_ln<<<NTOK / 8, 256, 0, stream>>>(bufX1, ln2g, ln2b, bufH, 128, SD);
  k_qkvp<<<dim3(L, 6), 128, 0, stream>>>(bufH, cWqT, cWkT, cWvT, cbq, cbk, cbv,
                                         cPj16, bufQp, bufKp, bufV);
  k_perf<<<dim3(L, 4), 128, 0, stream>>>(bufQp, bufKp, bufV, bufAttn, 1, 384);
  k_wo<<<NTOK / 64, 128, 0, stream>>>(bufAttn, cWoT, cbo, bufX1, outp);

  // ---- FeedForward residual (x2 lives in d_out) ----
  k_ln<<<NTOK / 8, 256, 0, stream>>>(outp, ln3g, ln3b, bufH, 128, SD);
  k_ffn<<<NTOK / 32, 64, 0, stream>>>(bufH, W1T, fb1, W2T, fb2, outp, outp);
}
